// Attention_19009525252123
// MI455X (gfx1250) — compile-verified, matrix operand order checked by reading, not run on hardware
//
#include <hip/hip_runtime.h>

#define HIDDEN 1152
#define NHEADS 16
#define HDIM   72
#define DPAD   96
#define BATCH  16
#define SEQ    1024
#define MTOT   (BATCH * SEQ)   /* 16384 */
#define N_QKV  (3 * HIDDEN)    /* 3456 */

typedef unsigned short u16;
typedef unsigned int   u32;

typedef __attribute__((ext_vector_type(16))) __bf16 bf16x16;
typedef __attribute__((ext_vector_type(8)))  float  f32x8;
typedef __attribute__((ext_vector_type(4)))  u32    u32x4;

// ---------- small helpers ----------
__device__ __forceinline__ u16 f2bf(float f) {
  union { float f; u32 u; } v; v.f = f;
  u32 u = v.u + 0x7FFFu + ((v.u >> 16) & 1u);   // round-to-nearest-even
  return (u16)(u >> 16);
}
__device__ __forceinline__ float bf2f(u16 h) {
  union { u32 u; float f; } v; v.u = ((u32)h) << 16; return v.f;
}

union FragU {
  u32x4   q[2];
  u16     s[16];
  bf16x16 v;
};

__device__ __forceinline__ f32x8 wmma_bf16(bf16x16 a, bf16x16 b, f32x8 c) {
  return __builtin_amdgcn_wmma_f32_16x16x32_bf16(false, a, false, b, (short)0, c,
                                                 false, false);
}

// A/B fragment for 16x16x32 bf16 WMMA from a row-major (K contiguous) bf16 row.
// Lane l<16: row l&15, K = {kk..kk+7, kk+16..kk+23}; lanes>=16 shift K by +8.
__device__ __forceinline__ bf16x16 frag_bf16_row(const u16* __restrict__ row,
                                                 int kk, int lane) {
  const int off = (lane & 16) ? 8 : 0;
  FragU u;
  u.q[0] = *reinterpret_cast<const u32x4*>(row + kk + off);
  u.q[1] = *reinterpret_cast<const u32x4*>(row + kk + off + 16);
  return u.v;
}

// ---------- kernel 1: fp32 -> bf16 conversion of weights and x ----------
__global__ void convert_all_kernel(const float* __restrict__ wq,
                                   const float* __restrict__ wp,
                                   const float* __restrict__ x,
                                   u16* __restrict__ wqb, u16* __restrict__ wpb,
                                   u16* __restrict__ xb) {
  const int N1 = N_QKV * HIDDEN;           // 3,981,312
  const int N2 = N1 + HIDDEN * HIDDEN;     // 5,308,416
  int i = blockIdx.x * blockDim.x + threadIdx.x;
  if (i < N1)      wqb[i]      = f2bf(wq[i]);
  else if (i < N2) wpb[i - N1] = f2bf(wp[i - N1]);
  else             xb[i - N2]  = f2bf(x[i - N2]);   // i-N2 < 16384*1152 (exact grid)
}

// ---------- kernel 2: zero padding region of q/k (d in [72,96)) ----------
__global__ void zeropad_kernel(u16* __restrict__ qkbuf) {
  int idx = blockIdx.x * blockDim.x + threadIdx.x;   // 2*256*1024*24 threads
  int row = idx / 24;
  int d = HDIM + (idx - row * 24);
  qkbuf[(size_t)row * DPAD + d] = 0;
}

// ---------- ping-pong software-pipelined 16m x 64n GEMM core, K=1152 ----------
// Unrolled by 2 so the two fragment buffers swap roles each half-step: load
// destinations are the loop-carried values themselves (no register rotation).
__device__ __forceinline__ void gemm_core(const u16* __restrict__ arow,
                                          const u16* const* brow,
                                          f32x8 acc[4], int lane) {
  bf16x16 a0 = frag_bf16_row(arow, 0, lane);
  bf16x16 b0[4];
#pragma unroll
  for (int t = 0; t < 4; ++t) b0[t] = frag_bf16_row(brow[t], 0, lane);

  for (int kk = 0; kk < HIDDEN; kk += 64) {          // 1152 = 18 * 64
    const int k1 = kk + 32;                          // always in-bounds
    const int k2 = (kk + 64 < HIDDEN) ? kk + 64 : kk;  // clamp on last iter
    __builtin_prefetch(arow + kk + 256, 0, 3);       // global_prefetch_b8

    bf16x16 a1 = frag_bf16_row(arow, k1, lane);
    bf16x16 b1[4];
#pragma unroll
    for (int t = 0; t < 4; ++t) b1[t] = frag_bf16_row(brow[t], k1, lane);
#pragma unroll
    for (int t = 0; t < 4; ++t) acc[t] = wmma_bf16(a0, b0[t], acc[t]);

    a0 = frag_bf16_row(arow, k2, lane);
#pragma unroll
    for (int t = 0; t < 4; ++t) b0[t] = frag_bf16_row(brow[t], k2, lane);
#pragma unroll
    for (int t = 0; t < 4; ++t) acc[t] = wmma_bf16(a1, b1[t], acc[t]);
  }
}

// ---------- kernel 3: QKV GEMM  C[16384,3456] = X * Wqkv^T + b ----------
// One wave computes a 16(m) x 64(n) tile. Q/K go to padded [which,b,h,s,96]
// bf16 buffer; V is written transposed: Vt[(b,h), d, s].
__global__ __launch_bounds__(128) void qkv_gemm_kernel(
    const u16* __restrict__ xb, const u16* __restrict__ wq,
    const float* __restrict__ bqkv, u16* __restrict__ qkbuf,
    u16* __restrict__ vt) {
  const int lane = threadIdx.x & 31;
  const int wid  = blockIdx.x * 4 + (threadIdx.x >> 5);
  const int NU = N_QKV / 64;                   // 54
  const int mt = wid / NU;
  const int nu = wid - mt * NU;
  const int rl = lane & 15;

  const u16* arow = xb + (size_t)(mt * 16 + rl) * HIDDEN;
  const u16* brow[4];
#pragma unroll
  for (int t = 0; t < 4; ++t)
    brow[t] = wq + (size_t)(nu * 64 + t * 16 + rl) * HIDDEN;

  f32x8 acc[4] = {};
  gemm_core(arow, brow, acc, lane);

  const int rbase = (lane & 16) ? 8 : 0;
#pragma unroll
  for (int t = 0; t < 4; ++t) {
    int nG = nu * 64 + t * 16 + rl;
    int which = nG / HIDDEN;
    int rem = nG - which * HIDDEN;
    int head = rem / HDIM;
    int d = rem - head * HDIM;
    float bias = bqkv[nG];
#pragma unroll
    for (int r = 0; r < 8; ++r) {
      int m = mt * 16 + rbase + r;
      int b = m >> 10, s = m & 1023;
      u16 hv = f2bf(acc[t][r] + bias);
      if (which < 2) {
        qkbuf[((size_t)((which * BATCH + b) * NHEADS + head) * SEQ + s) * DPAD + d] = hv;
      } else {
        vt[((size_t)(b * NHEADS + head) * DPAD + d) * SEQ + s] = hv;
      }
    }
  }
}

// ---------- kernel 4: RMSNorm over head_dim for q and k (in place) ----------
__global__ void rmsnorm_kernel(u16* __restrict__ qkbuf,
                               const float* __restrict__ qg,
                               const float* __restrict__ kg) {
  int idx = blockIdx.x * blockDim.x + threadIdx.x;  // 2*256*1024 rows
  const float* g = (idx >= 256 * 1024) ? kg : qg;
  u16* row = qkbuf + (size_t)idx * DPAD;
  float ss = 0.f;
  for (int d = 0; d < HDIM; ++d) { float v = bf2f(row[d]); ss += v * v; }
  float inv = rsqrtf(ss * (1.0f / HDIM) + 1.1920928955078125e-7f);
  for (int d = 0; d < HDIM; ++d) row[d] = f2bf(bf2f(row[d]) * inv * g[d]);
}

// ---------- kernel 5: flash attention, one wave per 16-row Q tile ----------
__global__ __launch_bounds__(128) void attn_kernel(
    const u16* __restrict__ qkbuf, const u16* __restrict__ vt,
    u16* __restrict__ attnbuf) {
  __shared__ u16 Pt[4][16][32];                 // per-wave P transpose tile
  const int lane = threadIdx.x & 31;
  const int wslot = threadIdx.x >> 5;
  const int wid = blockIdx.x * 4 + wslot;
  const int bh = wid >> 6;                      // 0..255 = b*16+h
  const int qt = wid & 63;                      // q tile within sequence
  const int rl = lane & 15;
  const int rbase = (lane & 16) ? 8 : 0;

  const u16* qrow  = qkbuf + ((size_t)bh * SEQ + qt * 16 + rl) * DPAD;
  const u16* kbase = qkbuf + ((size_t)(256 + bh) * SEQ) * DPAD;
  const u16* vbase = vt + (size_t)bh * DPAD * SEQ;

  bf16x16 qf[3];
#pragma unroll
  for (int i = 0; i < 3; ++i) qf[i] = frag_bf16_row(qrow, 32 * i, lane);

  f32x8 outacc[5] = {};
  float mrow[8], lrow[8];
#pragma unroll
  for (int r = 0; r < 8; ++r) { mrow[r] = -1e30f; lrow[r] = 0.f; }

  const float scale = 0.11785113019775793f;     // 1/sqrt(72)

  for (int j = 0; j < 32; ++j) {
    const int key0 = j * 32;
    f32x8 c0 = {}, c1 = {};
    const u16* kr0 = kbase + (size_t)(key0 + rl) * DPAD;
    const u16* kr1 = kr0 + (size_t)16 * DPAD;
#pragma unroll
    for (int i = 0; i < 3; ++i) {
      c0 = wmma_bf16(qf[i], frag_bf16_row(kr0, 32 * i, lane), c0);
      c1 = wmma_bf16(qf[i], frag_bf16_row(kr1, 32 * i, lane), c1);
    }

    // V fragments issued now: loads complete under the softmax VALU work
    bf16x16 vf[5];
#pragma unroll
    for (int t = 0; t < 5; ++t)
      vf[t] = frag_bf16_row(vbase + (size_t)(t * 16 + rl) * SEQ + key0, 0, lane);

    float alpha[8];
#pragma unroll
    for (int r = 0; r < 8; ++r) {
      float s0 = c0[r] * scale, s1 = c1[r] * scale;
      float t = fmaxf(s0, s1);
#pragma unroll
      for (int msk = 1; msk < 16; msk <<= 1) t = fmaxf(t, __shfl_xor(t, msk, 32));
      float mn = fmaxf(mrow[r], t);
      float a  = __expf(mrow[r] - mn);
      float p0 = __expf(s0 - mn), p1 = __expf(s1 - mn);
      float ps = p0 + p1;
#pragma unroll
      for (int msk = 1; msk < 16; msk <<= 1) ps += __shfl_xor(ps, msk, 32);
      lrow[r] = lrow[r] * a + ps;
      mrow[r] = mn;
      alpha[r] = a;
      c0[r] = p0; c1[r] = p1;
    }
#pragma unroll
    for (int t = 0; t < 5; ++t)
#pragma unroll
      for (int r = 0; r < 8; ++r) outacc[t][r] *= alpha[r];

    // transpose P (C layout: lane=col) -> A layout (lane=row) via LDS
#pragma unroll
    for (int r = 0; r < 8; ++r) {
      Pt[wslot][rbase + r][rl]      = f2bf(c0[r]);
      Pt[wslot][rbase + r][rl + 16] = f2bf(c1[r]);
    }
    __syncthreads();
    bf16x16 pf;
    {
      const u16* prow = &Pt[wslot][rl][0];
      const int off = (lane & 16) ? 8 : 0;
      FragU u;
      u.q[0] = *reinterpret_cast<const u32x4*>(prow + off);
      u.q[1] = *reinterpret_cast<const u32x4*>(prow + off + 16);
      pf = u.v;
    }
#pragma unroll
    for (int t = 0; t < 5; ++t)
      outacc[t] = wmma_bf16(pf, vf[t], outacc[t]);
    __syncthreads();
  }

  const int b = bh >> 4, h = bh & 15;
#pragma unroll
  for (int t = 0; t < 5; ++t) {
    int dim = t * 16 + rl;
    if (dim < HDIM) {
#pragma unroll
      for (int r = 0; r < 8; ++r) {
        int q = qt * 16 + rbase + r;
        attnbuf[((size_t)(b * SEQ + q)) * HIDDEN + h * HDIM + dim] =
            f2bf(outacc[t][r] / lrow[r]);
      }
    }
  }
}

// ---------- kernel 6: output projection, out = attn * Wproj^T + b ----------
__global__ __launch_bounds__(128) void proj_gemm_kernel(
    const u16* __restrict__ attnbuf, const u16* __restrict__ wp,
    const float* __restrict__ bproj, float* __restrict__ out) {
  const int lane = threadIdx.x & 31;
  const int wid = blockIdx.x * 4 + (threadIdx.x >> 5);
  const int NU = HIDDEN / 64;                   // 18
  const int mt = wid / NU;
  const int nu = wid - mt * NU;
  const int rl = lane & 15;

  const u16* arow = attnbuf + (size_t)(mt * 16 + rl) * HIDDEN;
  const u16* brow[4];
#pragma unroll
  for (int t = 0; t < 4; ++t)
    brow[t] = wp + (size_t)(nu * 64 + t * 16 + rl) * HIDDEN;

  f32x8 acc[4] = {};
  gemm_core(arow, brow, acc, lane);

  const int rbase = (lane & 16) ? 8 : 0;
#pragma unroll
  for (int t = 0; t < 4; ++t) {
    int nG = nu * 64 + t * 16 + rl;
    float bias = bproj[nG];
#pragma unroll
    for (int r = 0; r < 8; ++r) {
      int m = mt * 16 + rbase + r;
      out[(size_t)m * HIDDEN + nG] = acc[t][r] + bias;
    }
  }
}

// ---------- launcher ----------
extern "C" void kernel_launch(void* const* d_in, const int* in_sizes, int n_in,
                              void* d_out, int out_size, void* d_ws,
                              size_t ws_size, hipStream_t stream) {
  (void)in_sizes; (void)n_in; (void)out_size; (void)ws_size;
  const float* x     = (const float*)d_in[0];
  const float* wqkv  = (const float*)d_in[1];
  const float* bqkv  = (const float*)d_in[2];
  const float* qg    = (const float*)d_in[3];
  const float* kg    = (const float*)d_in[4];
  const float* wproj = (const float*)d_in[5];
  const float* bproj = (const float*)d_in[6];
  float* out = (float*)d_out;

  char* ws = (char*)d_ws;
  u16* wqb   = (u16*)(ws);                      //  7,962,624 B
  u16* wpb   = (u16*)(ws + 7962624);            //  2,654,208 B
  u16* qkbuf = (u16*)(ws + 10616832);           // 100,663,296 B (q,k padded)
  u16* vt    = (u16*)(ws + 111280128);          // 50,331,648 B (v transposed)
  u16* attnb = (u16*)(ws + 161611776);          // 37,748,736 B (attn out)
  u16* xbf   = attnb;  // bf16(x) aliases attnbuf: consumed by QKV GEMM before
                       // the attention kernel overwrites this region.

  convert_all_kernel<<<94464, 256, 0, stream>>>(wqkv, wproj, x, wqb, wpb, xbf);
  zeropad_kernel<<<49152, 256, 0, stream>>>(qkbuf);
  qkv_gemm_kernel<<<13824, 128, 0, stream>>>(xbf, wqb, bqkv, qkbuf, vt);
  rmsnorm_kernel<<<2048, 256, 0, stream>>>(qkbuf, qg, kg);
  attn_kernel<<<4096, 128, 0, stream>>>(qkbuf, vt, attnb);
  proj_gemm_kernel<<<4608, 128, 0, stream>>>(attnb, wpb, bproj, out);
}
